// Encoder_28595892256995
// MI455X (gfx1250) — matrix operand order repaired and checked by reading, not compiled, not run on hardware
//
// CDNA5 / gfx1250 tree-encoder.
// - 32 weight-pack kernels convert f32 weights -> f16 WMMA B-fragment layout in ws.
// - leaf kernel + 12 level kernels (serial launches on stream = inter-level sync).
// - Activations f16 in ws (~66MB) + packed weights (~16MB): all L2-resident (192MB).
// - All matmuls via v_wmma_f32_16x16x32_f16 (f32 accumulate).
#include <hip/hip_runtime.h>

typedef __attribute__((ext_vector_type(16))) _Float16 v16h;
typedef __attribute__((ext_vector_type(8)))  _Float16 v8h;
typedef __attribute__((ext_vector_type(8)))  float    v8f;
typedef _Float16 f16;

#define NDIRS 6
#define BATCHSZ 256
#define NPTS 4096

// ---- pack f32 (ndirs, Ksrc, OD) weights into f16 WMMA B-fragment order ----
// dst[(((dir*NKK + kkIdx)*OD + n)*2 + h)*16 + j] = (k<Ksrc)? src[(dir*Ksrc+k)*OD+n] : 0
//   with k = kkIdx*32 + h*16 + j,  NKK = KPad/32.
__global__ void pack_kernel(const float* __restrict__ src, f16* __restrict__ dst,
                            int Ksrc, int KPad, int OD, int ndirs) {
  int total = ndirs * KPad * OD;
  int i = blockIdx.x * blockDim.x + threadIdx.x;
  if (i >= total) return;
  int j    = i & 15;
  int h    = (i >> 4) & 1;
  int rem  = i >> 5;              // (dir*NKK + kkIdx)*OD + n
  int n    = rem % OD;
  int rem2 = rem / OD;
  int NKK  = KPad >> 5;
  int kkIdx = rem2 % NKK;
  int dir   = rem2 / NKK;
  int k = kkIdx*32 + h*16 + j;
  float v = (k < Ksrc) ? src[((size_t)dir*Ksrc + k)*OD + n] : 0.f;
  dst[i] = (f16)v;
}

// ---------------- leaf: (B*N,3) @ (3,8) + b, prelu -> f16 ----------------
__global__ void leaf_kernel(const float* __restrict__ pts,
                            const float* __restrict__ W,
                            const float* __restrict__ bias,
                            const float* __restrict__ a,
                            f16* __restrict__ act0, int total) {
  int idx = blockIdx.x * blockDim.x + threadIdx.x;
  if (idx >= total) return;
  float p0 = pts[3*idx+0], p1 = pts[3*idx+1], p2 = pts[3*idx+2];
  float av = a[0];
  v8h o;
#pragma unroll
  for (int c = 0; c < 8; ++c) {
    float y = fmaf(p0, W[c], fmaf(p1, W[8+c], fmaf(p2, W[16+c], bias[c])));
    y = (y >= 0.f) ? y : av * y;
    o[c] = (f16)y;
  }
  *(v8h*)&act0[8*idx] = o;
}

// A-fragment loader: two 16B LDS loads, contiguous k = 16h + s (matches packed B).
__device__ __forceinline__ v16h load_afrag(const f16* row, int h) {
  v8h lo = *(const v8h*)(row + h*16);
  v8h hi = *(const v8h*)(row + h*16 + 8);
  return __builtin_shufflevector(lo, hi, 0,1,2,3,4,5,6,7,8,9,10,11,12,13,14,15);
}

// ---------------- one tree level ----------------
template<int M, int IDIM, int ODIM, int SD, bool FINAL>
__global__ void __launch_bounds__(32*(ODIM/16))
level_kernel(const f16* __restrict__ prev,     // (B, 2M, IDIM)
             const f16* __restrict__ back3,    // (B, 8M, SD)   (SD>0 only)
             f16*       __restrict__ outAct,   // (B, M, ODIM)
             float*     __restrict__ outF32,   // final output (FINAL only)
             const int* __restrict__ vec,      // (B, M)
             const int* __restrict__ cl, const int* __restrict__ cr,
             const int* __restrict__ cs,
             const f16* __restrict__ WmP,      // packed (6, KP, ODIM)
             const float* __restrict__ am, const float* __restrict__ bm,
             const f16* __restrict__ WsP,      // packed (K2P, ODIM)
             const float* __restrict__ bs, const float* __restrict__ als,
             const f16* __restrict__ WmsP,     // packed (K3, ODIM)
             const float* __restrict__ bms, const float* __restrict__ alms,
             const int* __restrict__ drev, const int* __restrict__ dmap) {
  constexpr int K    = 2*IDIM;
  constexpr int KP   = (K < 32) ? 32 : K;
  constexpr int K3   = 2*ODIM;
  constexpr int XSLEN = (SD > 0 && K3 > KP) ? K3 : KP;
  constexpr int K2P  = (SD > 0) ? ((SD < 32) ? 32 : SD) : 0;
  constexpr int NW   = ODIM/16;
  constexpr int NT   = 32*NW;
  constexpr int NKK  = KP/32;

  __shared__ __align__(16) f16 xs[16*XSLEN];          // A-tile; reused for [out||smp]
  __shared__ __align__(16) f16 a2[(K2P > 0) ? 16*K2P : 16];
  __shared__ int vt[16];

  const int tid   = threadIdx.x;
  const int tile0 = blockIdx.x * 16;

  if (tid < 16) {
    int gr = tile0 + tid;
    int b = gr / M, m = gr - b*M;
    vt[tid] = dmap[vec[b*M + m]];
  }
  // Build x = [child_a || child_b] (swap by drev[vec]); 16B chunks, zero-pad K..KP.
  for (int e = tid; e < 16*(KP/8); e += NT) {
    int r  = e / (KP/8);
    int c0 = (e - r*(KP/8)) * 8;
    uint4 val = {0u,0u,0u,0u};
    if (c0 < K) {
      int gr = tile0 + r;
      int b = gr / M, m = gr - b*M;
      int dr = drev[vec[b*M + m]];
      int slot = (c0 < IDIM) ? 0 : 1;                 // 0 => first half
      int cc = c0 - slot*IDIM;
      int node = (dr ^ slot) ? cr[m] : cl[m];
      val = *(const uint4*)&prev[((size_t)b*(2*M) + node)*IDIM + cc];
    }
    *(uint4*)&xs[r*KP + c0] = val;
  }
  __syncthreads();

  const int lane = tid & 31;
  const int w    = tid >> 5;
  const int h    = lane >> 4;
  const int rl   = lane & 15;
  const int n    = 16*w + rl;

  // ---- main matmul: A shared across 6 direction matrices ----
  v8f acc[NDIRS];
#pragma unroll
  for (int d = 0; d < NDIRS; ++d) acc[d] = (v8f){0.f,0.f,0.f,0.f,0.f,0.f,0.f,0.f};
  const v16h* BpM = (const v16h*)WmP;
  for (int kkIdx = 0; kkIdx < NKK; ++kkIdx) {
    v16h af = load_afrag(&xs[rl*KP + kkIdx*32], h);
#pragma unroll
    for (int d = 0; d < NDIRS; ++d) {
      v16h bf = BpM[((size_t)(d*NKK + kkIdx)*ODIM + n)*2 + h];
      acc[d] = __builtin_amdgcn_wmma_f32_16x16x32_f16(false, af, false, bf,
                                                      (short)0, acc[d], false, false);
    }
  }
  v8f sel = {0.f,0.f,0.f,0.f,0.f,0.f,0.f,0.f};
#pragma unroll
  for (int d = 0; d < NDIRS; ++d) {
    float amv = am[d];
    float bmv = bm[d*ODIM + n];
#pragma unroll
    for (int i = 0; i < 8; ++i) {                     // C/D: vgpr i -> row i+8h
      float y = acc[d][i] + bmv;
      y = (y >= 0.f) ? y : amv*y;
      if (vt[i + 8*h] == d) sel[i] = y;
    }
  }

  if constexpr (SD == 0) {
#pragma unroll
    for (int i = 0; i < 8; ++i) {
      int gr = tile0 + i + 8*h;
      outAct[(size_t)gr*ODIM + n] = (f16)sel[i];
    }
  } else {
    __syncthreads();                                  // xs reads done everywhere
    // skip-path A-tile from level t-3 activations
    for (int e = tid; e < 16*(K2P/8); e += NT) {
      int r  = e / (K2P/8);
      int c0 = (e - r*(K2P/8)) * 8;
      uint4 val = {0u,0u,0u,0u};
      if (c0 < SD) {
        int gr = tile0 + r;
        int b = gr / M, m = gr - b*M;
        val = *(const uint4*)&back3[((size_t)b*(8*M) + cs[m])*SD + c0];
      }
      *(uint4*)&a2[r*K2P + c0] = val;
    }
    __syncthreads();

    constexpr int NK2 = K2P/32;
    v8f acc2 = {0.f,0.f,0.f,0.f,0.f,0.f,0.f,0.f};
    const v16h* BpS = (const v16h*)WsP;
    for (int kkIdx = 0; kkIdx < NK2; ++kkIdx) {
      v16h af = load_afrag(&a2[rl*K2P + kkIdx*32], h);
      v16h bf = BpS[((size_t)kkIdx*ODIM + n)*2 + h];
      acc2 = __builtin_amdgcn_wmma_f32_16x16x32_f16(false, af, false, bf,
                                                    (short)0, acc2, false, false);
    }
    float alsv = als[0], bsv = bs[n];
    float smp[8];
#pragma unroll
    for (int i = 0; i < 8; ++i) {
      float y = acc2[i] + bsv;
      smp[i] = (y >= 0.f) ? y : alsv*y;
    }
    // write [out || smp] merge A-tile (reuse xs)
#pragma unroll
    for (int i = 0; i < 8; ++i) {
      int r = i + 8*h;
      xs[r*K3 + n]        = (f16)sel[i];
      xs[r*K3 + ODIM + n] = (f16)smp[i];
    }
    __syncthreads();

    constexpr int NK3 = K3/32;
    v8f acc3 = {0.f,0.f,0.f,0.f,0.f,0.f,0.f,0.f};
    const v16h* BpX = (const v16h*)WmsP;
    for (int kkIdx = 0; kkIdx < NK3; ++kkIdx) {
      v16h af = load_afrag(&xs[rl*K3 + kkIdx*32], h);
      v16h bf = BpX[((size_t)kkIdx*ODIM + n)*2 + h];
      acc3 = __builtin_amdgcn_wmma_f32_16x16x32_f16(false, af, false, bf,
                                                    (short)0, acc3, false, false);
    }
    float almsv = alms[0], bmsv = bms[n];
#pragma unroll
    for (int i = 0; i < 8; ++i) {
      float y = acc3[i] + bmsv;
      y = (y >= 0.f) ? y : almsv*y;
      int gr = tile0 + i + 8*h;
      outAct[(size_t)gr*ODIM + n] = (f16)y;
      if constexpr (FINAL) outF32[(size_t)gr*ODIM + n] = y;
    }
  }
}

// ---------------- host ----------------
extern "C" void kernel_launch(void* const* d_in, const int* in_sizes, int n_in,
                              void* d_out, int out_size, void* d_ws, size_t ws_size,
                              hipStream_t stream) {
  (void)n_in; (void)out_size; (void)ws_size;
  static const int MsL[12]  = {2048,1024,512,256,128,64,32,16,8,4,2,1};
  static const int IdL[12]  = {8,16,16,32,32,64,64,128,128,256,256,512};
  static const int OdL[12]  = {16,16,32,32,64,64,128,128,256,256,512,512};
  static const int SdL[12]  = {0,0,8,16,16,32,32,64,64,128,128,256};

  const float* points = (const float*)d_in[0];
  const int* vecs[12];
  for (int t = 0; t < 12; ++t) vecs[t] = (const int*)d_in[1 + t];

  // Nested pytree flatten order probe: sorted keys (jax) => leaf {W,a,b}
  // (in_sizes[14]==1); insertion order => leaf {W,b,a} (in_sizes[14]==8).
  const bool sorted_order = (in_sizes[14] == 1);
  const float* leafW = (const float*)d_in[13];
  const float* leafA = (const float*)d_in[sorted_order ? 14 : 15];
  const float* leafB = (const float*)d_in[sorted_order ? 15 : 14];

  struct LvP { const float *Wm,*am,*bm,*Ws,*bs,*als,*Wms,*bms,*alms; };
  LvP lp[12] = {};
  int idx = 16;
  for (int t = 0; t < 12; ++t) {
    const bool skip = (t >= 2);
    if (!skip) {
      lp[t].Wm = (const float*)d_in[idx + 0];
      if (sorted_order) { lp[t].am = (const float*)d_in[idx+1]; lp[t].bm = (const float*)d_in[idx+2]; }
      else              { lp[t].bm = (const float*)d_in[idx+1]; lp[t].am = (const float*)d_in[idx+2]; }
      idx += 3;
    } else if (sorted_order) { // Wm, Wms, Ws, alms, als, am, bm, bms, bs
      lp[t].Wm  = (const float*)d_in[idx+0]; lp[t].Wms = (const float*)d_in[idx+1];
      lp[t].Ws  = (const float*)d_in[idx+2]; lp[t].alms= (const float*)d_in[idx+3];
      lp[t].als = (const float*)d_in[idx+4]; lp[t].am  = (const float*)d_in[idx+5];
      lp[t].bm  = (const float*)d_in[idx+6]; lp[t].bms = (const float*)d_in[idx+7];
      lp[t].bs  = (const float*)d_in[idx+8]; idx += 9;
    } else {                   // Wm, bm, am, Ws, bs, als, Wms, bms, alms
      lp[t].Wm  = (const float*)d_in[idx+0]; lp[t].bm  = (const float*)d_in[idx+1];
      lp[t].am  = (const float*)d_in[idx+2]; lp[t].Ws  = (const float*)d_in[idx+3];
      lp[t].bs  = (const float*)d_in[idx+4]; lp[t].als = (const float*)d_in[idx+5];
      lp[t].Wms = (const float*)d_in[idx+6]; lp[t].bms = (const float*)d_in[idx+7];
      lp[t].alms= (const float*)d_in[idx+8]; idx += 9;
    }
  }
  const int *cl[12], *cr[12], *cs[12];
  for (int t = 0; t < 12; ++t) {
    cl[t] = (const int*)d_in[idx]; cr[t] = (const int*)d_in[idx+1]; idx += 2;
    if (t >= 2) { cs[t] = (const int*)d_in[idx]; idx += 1; } else cs[t] = nullptr;
  }
  const int* drev = (const int*)d_in[idx++];
  const int* dmap = (const int*)d_in[idx++];

  // ws: f16 activation buffers (buf[l] = output of level l; l=0 leaf), then packed weights.
  static const int MsA[13] = {4096,2048,1024,512,256,128,64,32,16,8,4,2,1};
  static const int DsA[13] = {8,16,16,32,32,64,64,128,128,256,256,512,512};
  f16* base = (f16*)d_ws;
  size_t off = 0;
  f16* act[13];
  for (int l = 0; l < 13; ++l) { act[l] = base + off; off += (size_t)BATCHSZ*MsA[l]*DsA[l]; }
  f16 *wmP[12], *wsP[12], *wmsP[12];
  for (int t = 0; t < 12; ++t) {
    int K  = 2*IdL[t], KPad = (K < 32) ? 32 : K, OD = OdL[t];
    wmP[t] = base + off; off += (size_t)NDIRS*KPad*OD;
    if (SdL[t] > 0) {
      int K2P = (SdL[t] < 32) ? 32 : SdL[t];
      wsP[t]  = base + off; off += (size_t)K2P*OD;
      wmsP[t] = base + off; off += (size_t)2*OD*OD;
    } else { wsP[t] = nullptr; wmsP[t] = nullptr; }
  }

  // ---- pack weights (f32 -> f16 fragment layout) ----
  for (int t = 0; t < 12; ++t) {
    int K = 2*IdL[t], KPad = (K < 32) ? 32 : K, OD = OdL[t];
    int total = NDIRS*KPad*OD;
    pack_kernel<<<(total + 255)/256, 256, 0, stream>>>(lp[t].Wm, wmP[t], K, KPad, OD, NDIRS);
    if (SdL[t] > 0) {
      int K2P = (SdL[t] < 32) ? 32 : SdL[t];
      int tot2 = K2P*OD;
      pack_kernel<<<(tot2 + 255)/256, 256, 0, stream>>>(lp[t].Ws, wsP[t], SdL[t], K2P, OD, 1);
      int tot3 = 2*OD*OD;
      pack_kernel<<<(tot3 + 255)/256, 256, 0, stream>>>(lp[t].Wms, wmsP[t], 2*OD, 2*OD, OD, 1);
    }
  }

  leaf_kernel<<<(BATCHSZ*NPTS + 255)/256, 256, 0, stream>>>(
      points, leafW, leafB, leafA, act[0], BATCHSZ*NPTS);

#define LAUNCH(t, Mv, ID, OD, SDv, FIN)                                          \
  level_kernel<Mv, ID, OD, SDv, FIN>                                             \
    <<<(BATCHSZ*(Mv))/16, 32*((OD)/16), 0, stream>>>(                            \
      act[(t)-1], (SDv) > 0 ? act[(t) >= 3 ? (t)-3 : 0] : nullptr, act[t],       \
      (FIN) ? (float*)d_out : nullptr,                                           \
      vecs[(t)-1], cl[(t)-1], cr[(t)-1], cs[(t)-1],                              \
      wmP[(t)-1], lp[(t)-1].am, lp[(t)-1].bm,                                    \
      wsP[(t)-1], lp[(t)-1].bs, lp[(t)-1].als,                                   \
      wmsP[(t)-1], lp[(t)-1].bms, lp[(t)-1].alms, drev, dmap)

  LAUNCH(1,  2048,   8,  16,   0, false);
  LAUNCH(2,  1024,  16,  16,   0, false);
  LAUNCH(3,   512,  16,  32,   8, false);
  LAUNCH(4,   256,  32,  32,  16, false);
  LAUNCH(5,   128,  32,  64,  16, false);
  LAUNCH(6,    64,  64,  64,  32, false);
  LAUNCH(7,    32,  64, 128,  32, false);
  LAUNCH(8,    16, 128, 128,  64, false);
  LAUNCH(9,     8, 128, 256,  64, false);
  LAUNCH(10,    4, 256, 256, 128, false);
  LAUNCH(11,    2, 256, 512, 128, false);
  LAUNCH(12,    1, 512, 512, 256, true);
#undef LAUNCH
}
